// TransCBA_39109972197394
// MI455X (gfx1250) — matrix operand order repaired and checked by reading, not compiled, not run on hardware
//
#include <hip/hip_runtime.h>
#include <hip/hip_bf16.h>
#include <math.h>

// ---------------- Problem constants (from reference) ----------------
#define BATCH 8
#define CCH   256          // C
#define NHD   8            // heads
#define HDIM  32           // head dim for V
#define KDIM  16           // head dim for Q/K
#define HCH_  512          // C + 2*KD*NH
#define HH    32
#define WW    32
#define NPIX  1024         // H*W
static constexpr float SCALE_F = 0.25f;  // KD^-0.5

// ---------------- WMMA types (CDNA5 / gfx1250, wave32) ----------------
typedef __attribute__((ext_vector_type(16))) __bf16 v16bf;
typedef __attribute__((ext_vector_type(8)))  float  v8f;
typedef int v4i_ __attribute__((vector_size(16)));   // 16-byte int vector for async LDS builtin

union FragBF { v16bf v; unsigned short u[16]; };

__device__ __forceinline__ unsigned short f2bf(float f) {
  unsigned int x = __builtin_bit_cast(unsigned int, f);
  x += 0x7FFFu + ((x >> 16) & 1u);          // round-to-nearest-even
  return (unsigned short)(x >> 16);
}
__device__ __forceinline__ float bf2f(unsigned short h) {
  unsigned int x = ((unsigned int)h) << 16;
  return __builtin_bit_cast(float, x);
}

__device__ __forceinline__ v8f wmma_bf16(const FragBF& a, const FragBF& b, const v8f& c) {
  // D = A(16x32 bf16) * B(32x16 bf16) + C(16x16 f32)
  return __builtin_amdgcn_wmma_f32_16x16x32_bf16(
      /*neg_a=*/false, a.v, /*neg_b=*/false, b.v,
      /*c_mod=*/(short)0, c, /*reuse_a=*/false, /*reuse_b=*/false);
}

// -------- async global->LDS copy (gfx1250 GLOBAL_LOAD_ASYNC_TO_LDS_B128) --------
#if defined(__has_builtin)
#if __has_builtin(__builtin_amdgcn_global_load_async_to_lds_b128)
#define HAVE_ASYNC_LDS 1
#endif
#endif
#ifndef HAVE_ASYNC_LDS
#define HAVE_ASYNC_LDS 0
#endif

__device__ __forceinline__ void copy16_g2l(const unsigned short* __restrict__ g,
                                           unsigned short* __restrict__ l) {
#if HAVE_ASYNC_LDS
  void* gv = (void*)g;   // strip const; builtin takes non-const pointers
  void* lv = (void*)l;
  __builtin_amdgcn_global_load_async_to_lds_b128(
      (__attribute__((address_space(1))) v4i_*)gv,
      (__attribute__((address_space(3))) v4i_*)lv, /*offset=*/0, /*cpol=*/0);
#else
  *(uint4*)l = *(const uint4*)g;
#endif
}
__device__ __forceinline__ void async_wait_all() {
#if HAVE_ASYNC_LDS
#if defined(__has_builtin) && __has_builtin(__builtin_amdgcn_s_wait_asynccnt)
  __builtin_amdgcn_s_wait_asynccnt(0);
#else
  asm volatile("s_wait_asynccnt 0x0" ::: "memory");
#endif
#endif
}

// A fragment 16x32: row-major src, element (m,k) at src[(m0+m)*ld + k0+k].
// CDNA5 layout: lane L -> m=L&15; lanes0-15: V0..3 hold K=0..7 pairs, V4..7 K=16..23;
// lanes16-31: V0..3 K=8..15, V4..7 K=24..31. Row reads are contiguous -> b128.
__device__ __forceinline__ FragBF load_a_bf16(const unsigned short* src, int ld, int m0, int k0) {
  FragBF f; int lane = threadIdx.x & 31;
  int m = lane & 15, kb = (lane >> 4) * 8;
  const unsigned short* row = src + (size_t)(m0 + m) * ld + k0;
#pragma unroll
  for (int r = 0; r < 4; ++r) {
    f.u[2*r]   = row[kb + 2*r];
    f.u[2*r+1] = row[kb + 2*r + 1];
  }
#pragma unroll
  for (int r = 4; r < 8; ++r) {
    f.u[2*r]   = row[16 + kb + 2*(r-4)];
    f.u[2*r+1] = row[16 + kb + 2*(r-4) + 1];
  }
  return f;
}

// B fragment 32x16 from global, element (k,n) at src[(k0+k)*ld + n0+n].
// lane L -> n=L&15; lanes0-15 hold K=0..7,16..23, lanes16-31 K=8..15,24..31 (pairs per VGPR).
// klim allows zero-padding K (KD=16 case). Used only for once-per-wave Q loads.
__device__ __forceinline__ FragBF load_b_bf16(const unsigned short* src, int ld,
                                              int k0, int n0, int klim) {
  FragBF f; int lane = threadIdx.x & 31;
  int n = lane & 15, kb = (lane >> 4) * 8;
#pragma unroll
  for (int r = 0; r < 8; ++r) {
    int ka = kb + 2*r + (r >= 4 ? 8 : 0), kc = ka + 1;
    f.u[2*r]   = (ka < klim) ? src[(size_t)(k0 + ka) * ld + n0 + n] : (unsigned short)0;
    f.u[2*r+1] = (kc < klim) ? src[(size_t)(k0 + kc) * ld + n0 + n] : (unsigned short)0;
  }
  return f;
}

// B fragment from an LDS-staged [k][n] bf16 tile with row pitch `pitch`.
__device__ __forceinline__ FragBF load_b_lds(const unsigned short* Xs, int pitch, int n0) {
  FragBF f; int lane = threadIdx.x & 31;
  int n = n0 + (lane & 15), kb = (lane >> 4) * 8;
#pragma unroll
  for (int j = 0; j < 16; ++j) f.u[j] = Xs[((j & 8) * 2 + kb + (j & 7)) * pitch + n];
  return f;
}

// ---------------- elementwise cast ----------------
__global__ void k_cast_bf16(const float* __restrict__ src, unsigned short* __restrict__ dst, int n) {
  int i = blockIdx.x * blockDim.x + threadIdx.x;
  if (i < n) dst[i] = f2bf(src[i]);
}

// ---------------- QKV GEMM: Out[b,o,n] = sum_c W[o,c]*X[b,c,n] + bias[o]  (bf16 out) ----
// Block = 256 threads (8 waves) computing a 64(o) x 128(n) tile; the 32x128 X tile is
// staged into LDS (async global->LDS when available) and feeds 4 WMMAs per wave.
// grid(N/128, HCH/64, B)
__global__ void k_qkv_gemm(const unsigned short* __restrict__ W,   // (HCH, C) bf16
                           const unsigned short* __restrict__ X,   // (B, C, N) bf16
                           const float* __restrict__ bias,         // (HCH)
                           unsigned short* __restrict__ Out) {     // (B, HCH, N) bf16
  __shared__ unsigned short Xs[32 * 128];            // 8 KB: X[k][n] tile
  int wave = threadIdx.x >> 5, lane = threadIdx.x & 31;
  int nb = blockIdx.x * 128;
  int ob = blockIdx.y * 64;
  int b  = blockIdx.z;
  const unsigned short* Xb = X + (size_t)b * CCH * NPIX;

  // staging map: thread t copies 32 bytes of row kr at columns cg..cg+15
  int kr = threadIdx.x >> 3, cg = (threadIdx.x & 7) * 16;

  v8f acc[4] = {v8f{}, v8f{}, v8f{}, v8f{}};
  for (int k0 = 0; k0 < CCH; k0 += 32) {
    __syncthreads();                                 // previous tile consumed
    const unsigned short* src = Xb + (size_t)(k0 + kr) * NPIX + nb + cg;
    __builtin_prefetch(Xb + (size_t)(k0 + 32 + kr) * NPIX + nb + cg, 0, 0);
    copy16_g2l(src,     &Xs[kr * 128 + cg]);
    copy16_g2l(src + 8, &Xs[kr * 128 + cg + 8]);
    async_wait_all();
    __syncthreads();
    FragBF bb = load_b_lds(Xs, 128, wave * 16);
#pragma unroll
    for (int ot = 0; ot < 4; ++ot) {
      FragBF a = load_a_bf16(W, CCH, ob + ot * 16, k0);
      acc[ot] = wmma_bf16(a, bb, acc[ot]);
    }
  }
  int n = lane & 15, mb = (lane >> 4) * 8;
  unsigned short* outp = Out + (size_t)b * HCH_ * NPIX;
#pragma unroll
  for (int ot = 0; ot < 4; ++ot)
#pragma unroll
    for (int r = 0; r < 8; ++r) {
      int o = ob + ot * 16 + mb + r;
      outp[(size_t)o * NPIX + nb + wave * 16 + n] = f2bf(acc[ot][r] + bias[o]);
    }
}

// ---------------- PROJ GEMM: Out[b,o,n] = sum_c W[o,c]*X[b,c,n] + bias[o]  (f32 out) ----
// Same tiling as qkv GEMM; the f32 X tile is converted to bf16 while staging into LDS
// with packed 16-byte LDS stores. grid(N/128, C/64, B)
__global__ void k_proj_gemm(const unsigned short* __restrict__ W,  // (C, C) bf16
                            const float* __restrict__ X,           // (B, C, N) f32
                            const float* __restrict__ bias,        // (C)
                            float* __restrict__ Out) {             // (B, C, N) f32
  __shared__ unsigned short Xs[32 * 128];            // 8 KB
  int wave = threadIdx.x >> 5, lane = threadIdx.x & 31;
  int nb = blockIdx.x * 128;
  int ob = blockIdx.y * 64;
  int b  = blockIdx.z;
  const float* Xb = X + (size_t)b * CCH * NPIX;
  int kr = threadIdx.x >> 3, cg = (threadIdx.x & 7) * 16;

  v8f acc[4] = {v8f{}, v8f{}, v8f{}, v8f{}};
  for (int k0 = 0; k0 < CCH; k0 += 32) {
    __syncthreads();
    const float* src = Xb + (size_t)(k0 + kr) * NPIX + nb + cg;
    unsigned int t[8];
#pragma unroll
    for (int j = 0; j < 8; ++j)
      t[j] = (unsigned)f2bf(src[2*j]) | ((unsigned)f2bf(src[2*j + 1]) << 16);
    *(uint4*)&Xs[kr * 128 + cg]     = make_uint4(t[0], t[1], t[2], t[3]);
    *(uint4*)&Xs[kr * 128 + cg + 8] = make_uint4(t[4], t[5], t[6], t[7]);
    __syncthreads();
    FragBF bb = load_b_lds(Xs, 128, wave * 16);
#pragma unroll
    for (int ot = 0; ot < 4; ++ot) {
      FragBF a = load_a_bf16(W, CCH, ob + ot * 16, k0);
      acc[ot] = wmma_bf16(a, bb, acc[ot]);
    }
  }
  int n = lane & 15, mb = (lane >> 4) * 8;
  float* outp = Out + (size_t)b * CCH * NPIX;
#pragma unroll
  for (int ot = 0; ot < 4; ++ot)
#pragma unroll
    for (int r = 0; r < 8; ++r) {
      int o = ob + ot * 16 + mb + r;
      outp[(size_t)o * NPIX + nb + wave * 16 + n] = acc[ot][r] + bias[o];
    }
}

// ---------------- Fused flash cross-attention ----------------
// Att[b, h*HD+d, i] = SCALE * sum_j V[d,j] * softmax_j(Q^T K)[i,j]
// Block = 8 waves sharing one (b, head); wave w owns query tile qt=blockIdx.x*8+w.
// Per key tile, all 256 threads cooperatively transpose the 16x16 K tile into LDS
// ([j][kd] layout) with coalesced global reads; each wave's Kᵀ A-fragment is then a
// single aligned 16B LDS read. Scores are computed transposed (keys on the M/VGPR
// axis) so softmax-over-keys is a column reduction: 8 VGPRs in-lane + one
// shfl_xor(16) across lane halves. P is repacked in-register as the B operand of
// O = V*P from the lane's own p values. KD=16 is zero-padded to the K=32 instruction.
// grid(N/16/8, NH, B), block 256.
__global__ void k_cross_attn(const unsigned short* __restrict__ QKVq,
                             const unsigned short* __restrict__ QKVk,
                             const unsigned short* __restrict__ QKVv,
                             float* __restrict__ Att) {             // (B, C, N)
  __shared__ unsigned short Ks[16 * 16];              // Kᵀ tile: [j][kd], 512 B
  int wave = threadIdx.x >> 5, lane = threadIdx.x & 31;
  int qt = blockIdx.x * 8 + wave;
  int h = blockIdx.y, b = blockIdx.z;
  const unsigned short* qbase = QKVq + ((size_t)b * HCH_ + h * 64)          * NPIX;
  const unsigned short* kbase = QKVk + ((size_t)b * HCH_ + h * 64 + KDIM)   * NPIX;
  const unsigned short* vbase = QKVv + ((size_t)b * HCH_ + h * 64 + 2*KDIM) * NPIX;

  // Q B-fragment for this wave's query tile (K=kd, zero-padded 16..31); once.
  FragBF qf = load_b_bf16(qbase, NPIX, 0, qt * 16, KDIM);

  float Mi = -1e30f, Li = 0.f;
  v8f o0 = {}, o1 = {};
  int m  = lane & 15;
  int k8 = (lane >> 4) * 8;
  bool hi = (lane >= 16);
  // staging map: thread t -> (j = t&15, kd = t>>4); consecutive t = consecutive j
  int sj = threadIdx.x & 15, sk = threadIdx.x >> 4;

  for (int jt = 0; jt < 64; ++jt) {
    __syncthreads();                                  // previous Kᵀ tile consumed
    Ks[sj * 16 + sk] = kbase[(size_t)sk * NPIX + jt * 16 + sj];
    __syncthreads();

    // A = Kᵀ: A[m=j_local, kd]; 8 contiguous kd = one 16B LDS read; kd>=16 zero.
    FragBF kf;
    *(uint4*)&kf.u[0] = *(const uint4*)&Ks[m * 16 + k8];
#pragma unroll
    for (int r = 4; r < 8; ++r) { kf.u[2*r] = 0; kf.u[2*r+1] = 0; }

    v8f zero = {};
    v8f s = wmma_bf16(kf, qf, zero);                  // Sᵀ[j,i] logits

    // --- online softmax over j (column direction) ---
    float tm = s[0];
#pragma unroll
    for (int r = 1; r < 8; ++r) tm = fmaxf(tm, s[r]);
    tm = fmaxf(tm, __shfl_xor(tm, 16, 32));
    float Mn = fmaxf(Mi, tm);
    float sc = __expf(Mi - Mn);
    float p[8], ps = 0.f;
#pragma unroll
    for (int r = 0; r < 8; ++r) { p[r] = __expf(s[r] - Mn); ps += p[r]; }
    ps += __shfl_xor(ps, 16, 32);
    Li = Li * sc + ps;
    Mi = Mn;
#pragma unroll
    for (int r = 0; r < 8; ++r) { o0[r] *= sc; o1[r] *= sc; }

    // --- repack P (f32 D-layout) into bf16 B operand P[j,i], j padded 16..31 ---
    float pp[8];
#pragma unroll
    for (int r = 0; r < 8; ++r) pp[r] = __shfl_xor(p[r], 16, 32);
    FragBF pf;
#pragma unroll
    for (int r = 0; r < 8; ++r) {
      if (r < 4)       { pf.u[2*r] = f2bf(p[2*r]);     pf.u[2*r+1] = f2bf(p[2*r+1]); }
      else             { pf.u[2*r] = 0;                pf.u[2*r+1] = 0; }
    }

    // --- O += V * P  (two 16-row d tiles; rows contiguous -> b128 loads) ---
#pragma unroll
    for (int dt = 0; dt < 2; ++dt) {
      FragBF vf;
      const unsigned short* row = vbase + (size_t)(dt * 16 + m) * NPIX + jt * 16;
#pragma unroll
      for (int r = 0; r < 4; ++r) {
        vf.u[2*r]   = row[k8 + 2*r];
        vf.u[2*r+1] = row[k8 + 2*r + 1];
      }
#pragma unroll
      for (int r = 4; r < 8; ++r) { vf.u[2*r] = 0; vf.u[2*r+1] = 0; }
      if (dt == 0) o0 = wmma_bf16(vf, pf, o0);
      else         o1 = wmma_bf16(vf, pf, o1);
    }
  }

  // Epilogue: reference scales the *probabilities* by SCALE after softmax.
  float inv = SCALE_F / Li;
  int n = lane & 15, mb = (lane >> 4) * 8;
  float* dst = Att + ((size_t)b * CCH + h * HDIM) * NPIX + qt * 16 + n;
#pragma unroll
  for (int r = 0; r < 8; ++r) {
    dst[(size_t)(mb + r)      * NPIX] = o0[r] * inv;
    dst[(size_t)(16 + mb + r) * NPIX] = o1[r] * inv;
  }
}

// ---------------- depthwise 3x3 positional encoding, added in place ----------------
__global__ void k_dwconv_add(const unsigned short* __restrict__ QKV,  // v source (B,HCH,N) bf16
                             const float* __restrict__ W,             // (C,1,3,3)
                             const float* __restrict__ bias,          // (C)
                             float* __restrict__ Att) {               // (B,C,N), += result
  int idx = blockIdx.x * blockDim.x + threadIdx.x;
  if (idx >= BATCH * CCH * NPIX) return;
  int n = idx & (NPIX - 1);
  int c = (idx >> 10) & (CCH - 1);
  int b = idx >> 18;
  int x = n & (WW - 1), y = n >> 5;
  int head = c >> 5, d = c & 31;
  const unsigned short* v = QKV + ((size_t)b * HCH_ + head * 64 + 2*KDIM + d) * NPIX;
  const float* w = W + c * 9;
  float acc = bias[c];
#pragma unroll
  for (int ky = -1; ky <= 1; ++ky)
#pragma unroll
    for (int kx = -1; kx <= 1; ++kx) {
      int yy = y + ky, xx = x + kx;
      if (yy >= 0 && yy < HH && xx >= 0 && xx < WW)
        acc += w[(ky + 1) * 3 + (kx + 1)] * bf2f(v[yy * WW + xx]);
    }
  Att[idx] += acc;
}

// ---------------- spatial-attention pooling: avg/max over channels ----------------
__global__ void k_pool_spatial(const float* __restrict__ O, float* __restrict__ pooled) {
  int b = blockIdx.y;
  int n = blockIdx.x * blockDim.x + threadIdx.x;
  if (n >= NPIX) return;
  const float* p = O + (size_t)b * CCH * NPIX + n;
  float s = 0.f, m = -1e30f;
  for (int c = 0; c < CCH; ++c) { float v = p[(size_t)c * NPIX]; s += v; m = fmaxf(m, v); }
  pooled[((size_t)b * 2 + 0) * NPIX + n] = s * (1.0f / CCH);
  pooled[((size_t)b * 2 + 1) * NPIX + n] = m;
}

// ---------------- 7x7 conv over 2-channel pooled map + sigmoid ----------------
__global__ void k_sa_conv7(const float* __restrict__ pooled, const float* __restrict__ W,
                           const float* __restrict__ bias, float* __restrict__ SA) {
  int b = blockIdx.y;
  int n = blockIdx.x * blockDim.x + threadIdx.x;
  if (n >= NPIX) return;
  int x = n & 31, y = n >> 5;
  float acc = bias[0];
  for (int ic = 0; ic < 2; ++ic) {
    const float* src = pooled + ((size_t)b * 2 + ic) * NPIX;
    const float* w = W + ic * 49;
    for (int ky = -3; ky <= 3; ++ky)
      for (int kx = -3; kx <= 3; ++kx) {
        int yy = y + ky, xx = x + kx;
        if (yy >= 0 && yy < HH && xx >= 0 && xx < WW)
          acc += w[(ky + 3) * 7 + (kx + 3)] * src[yy * WW + xx];
      }
  }
  SA[(size_t)b * NPIX + n] = 1.0f / (1.0f + __expf(-acc));
}

// ---------------- channel attention (SE-style MLP gate) ----------------
__global__ void k_channel_attn(const float* __restrict__ O,
                               const float* __restrict__ w1, const float* __restrict__ b1,
                               const float* __restrict__ w2, const float* __restrict__ b2,
                               float* __restrict__ CA) {
  __shared__ float avg[CCH], mx[CCH], hA[16], hM[16];
  int b = blockIdx.x, c = threadIdx.x;                 // 256 threads
  const float* p = O + ((size_t)b * CCH + c) * NPIX;
  float s = 0.f, m = -1e30f;
  for (int n = 0; n < NPIX; ++n) { float v = p[n]; s += v; m = fmaxf(m, v); }
  avg[c] = s * (1.0f / NPIX); mx[c] = m;
  __syncthreads();
  if (c < 16) {
    float sa = b1[c], sm = b1[c];
    for (int k = 0; k < CCH; ++k) { float w = w1[c * CCH + k]; sa += w * avg[k]; sm += w * mx[k]; }
    hA[c] = fmaxf(sa, 0.f); hM[c] = fmaxf(sm, 0.f);
  }
  __syncthreads();
  float ga = b2[c], gm = b2[c];
#pragma unroll
  for (int k = 0; k < 16; ++k) { float w = w2[c * 16 + k]; ga += w * hA[k]; gm += w * hM[k]; }
  CA[(size_t)b * CCH + c] = 1.0f / (1.0f + __expf(-(ga + gm)));
}

// ---------------- final combine ----------------
__global__ void k_final(const float* __restrict__ rgb_o, const float* __restrict__ chm_o,
                        const float* __restrict__ rgb_sa, const float* __restrict__ chm_sa,
                        const float* __restrict__ rgb_ca, const float* __restrict__ chm_ca,
                        float* __restrict__ out) {
  int idx = blockIdx.x * blockDim.x + threadIdx.x;
  if (idx >= BATCH * CCH * NPIX) return;
  int n = idx & (NPIX - 1);
  int c = (idx >> 10) & (CCH - 1);
  int b = idx >> 18;
  out[idx] = rgb_o[idx] * chm_sa[b * NPIX + n] * chm_ca[b * CCH + c]
           + chm_o[idx] * rgb_sa[b * NPIX + n] * rgb_ca[b * CCH + c];
}

// ======================= host side =======================
extern "C" void kernel_launch(void* const* d_in, const int* in_sizes, int n_in,
                              void* d_out, int out_size, void* d_ws, size_t ws_size,
                              hipStream_t stream) {
  (void)in_sizes; (void)n_in; (void)out_size; (void)ws_size;
  const float* rgb        = (const float*)d_in[0];
  const float* chm        = (const float*)d_in[1];
  const float* rgb_qkv_w  = (const float*)d_in[2];
  const float* rgb_qkv_b  = (const float*)d_in[3];
  const float* chm_qkv_w  = (const float*)d_in[4];
  const float* chm_qkv_b  = (const float*)d_in[5];
  const float* rgb_pe_w   = (const float*)d_in[6];
  const float* rgb_pe_b   = (const float*)d_in[7];
  const float* chm_pe_w   = (const float*)d_in[8];
  const float* chm_pe_b   = (const float*)d_in[9];
  const float* rgb_proj_w = (const float*)d_in[10];
  const float* rgb_proj_b = (const float*)d_in[11];
  const float* chm_proj_w = (const float*)d_in[12];
  const float* chm_proj_b = (const float*)d_in[13];
  const float* rgb_fc1_w  = (const float*)d_in[14];
  const float* rgb_fc1_b  = (const float*)d_in[15];
  const float* rgb_fc2_w  = (const float*)d_in[16];
  const float* rgb_fc2_b  = (const float*)d_in[17];
  const float* chm_fc1_w  = (const float*)d_in[18];
  const float* chm_fc1_b  = (const float*)d_in[19];
  const float* chm_fc2_w  = (const float*)d_in[20];
  const float* chm_fc2_b  = (const float*)d_in[21];
  const float* rgb_sa_w   = (const float*)d_in[22];
  const float* rgb_sa_b   = (const float*)d_in[23];
  const float* chm_sa_w   = (const float*)d_in[24];
  const float* chm_sa_b   = (const float*)d_in[25];
  float* out = (float*)d_out;

  // --- workspace carve-up (256B aligned) ---
  char* ws = (char*)d_ws;
  size_t off = 0;
  auto alloc = [&](size_t bytes) -> char* {
    char* p = ws + off;
    off += (bytes + 255) & ~(size_t)255;
    return p;
  };
  const size_t BCN = (size_t)BATCH * CCH * NPIX;          // 2M elems
  const size_t BHN = (size_t)BATCH * HCH_ * NPIX;         // 4M elems
  unsigned short* Xrgb  = (unsigned short*)alloc(BCN * 2);
  unsigned short* Xchm  = (unsigned short*)alloc(BCN * 2);
  unsigned short* WqR   = (unsigned short*)alloc((size_t)HCH_ * CCH * 2);
  unsigned short* WqC   = (unsigned short*)alloc((size_t)HCH_ * CCH * 2);
  unsigned short* WpR   = (unsigned short*)alloc((size_t)CCH * CCH * 2);
  unsigned short* WpC   = (unsigned short*)alloc((size_t)CCH * CCH * 2);
  unsigned short* QKVr  = (unsigned short*)alloc(BHN * 2);
  unsigned short* QKVc  = (unsigned short*)alloc(BHN * 2);
  float* AttR  = (float*)alloc(BCN * 4);
  float* AttC  = (float*)alloc(BCN * 4);
  float* Orgb  = (float*)alloc(BCN * 4);
  float* Ochm  = (float*)alloc(BCN * 4);
  float* PoolR = (float*)alloc((size_t)BATCH * 2 * NPIX * 4);
  float* PoolC = (float*)alloc((size_t)BATCH * 2 * NPIX * 4);
  float* SaR   = (float*)alloc((size_t)BATCH * NPIX * 4);
  float* SaC   = (float*)alloc((size_t)BATCH * NPIX * 4);
  float* CaR   = (float*)alloc((size_t)BATCH * CCH * 4);
  float* CaC   = (float*)alloc((size_t)BATCH * CCH * 4);

  // 1) cast activations + GEMM weights to bf16
  k_cast_bf16<<<(int)((BCN + 255) / 256), 256, 0, stream>>>(rgb, Xrgb, (int)BCN);
  k_cast_bf16<<<(int)((BCN + 255) / 256), 256, 0, stream>>>(chm, Xchm, (int)BCN);
  k_cast_bf16<<<(HCH_ * CCH + 255) / 256, 256, 0, stream>>>(rgb_qkv_w, WqR, HCH_ * CCH);
  k_cast_bf16<<<(HCH_ * CCH + 255) / 256, 256, 0, stream>>>(chm_qkv_w, WqC, HCH_ * CCH);
  k_cast_bf16<<<(CCH * CCH + 255) / 256, 256, 0, stream>>>(rgb_proj_w, WpR, CCH * CCH);
  k_cast_bf16<<<(CCH * CCH + 255) / 256, 256, 0, stream>>>(chm_proj_w, WpC, CCH * CCH);

  // 2) QKV projections (WMMA bf16, LDS/async staged)
  dim3 gq(NPIX / 128, HCH_ / 64, BATCH);
  k_qkv_gemm<<<gq, 256, 0, stream>>>(WqR, Xrgb, rgb_qkv_b, QKVr);
  k_qkv_gemm<<<gq, 256, 0, stream>>>(WqC, Xchm, chm_qkv_b, QKVc);

  // 3) fused flash cross-attention (WMMA bf16, cooperative K staging)
  //    r_att: Q=rgb, K=chm, V=rgb ; c_att: Q=chm, K=rgb, V=chm
  dim3 ga(NPIX / 16 / 8, NHD, BATCH);
  k_cross_attn<<<ga, 256, 0, stream>>>(QKVr, QKVc, QKVr, AttR);
  k_cross_attn<<<ga, 256, 0, stream>>>(QKVc, QKVr, QKVc, AttC);

  // 4) depthwise 3x3 positional encoding added to attention output
  k_dwconv_add<<<(int)((BCN + 255) / 256), 256, 0, stream>>>(QKVr, rgb_pe_w, rgb_pe_b, AttR);
  k_dwconv_add<<<(int)((BCN + 255) / 256), 256, 0, stream>>>(QKVc, chm_pe_w, chm_pe_b, AttC);

  // 5) output projections (WMMA bf16, f32 out)
  dim3 gp(NPIX / 128, CCH / 64, BATCH);
  k_proj_gemm<<<gp, 256, 0, stream>>>(WpR, AttR, rgb_proj_b, Orgb);
  k_proj_gemm<<<gp, 256, 0, stream>>>(WpC, AttC, chm_proj_b, Ochm);

  // 6) spatial + channel attention gates
  dim3 gs(NPIX / 256, BATCH);
  k_pool_spatial<<<gs, 256, 0, stream>>>(Orgb, PoolR);
  k_pool_spatial<<<gs, 256, 0, stream>>>(Ochm, PoolC);
  k_sa_conv7<<<gs, 256, 0, stream>>>(PoolR, rgb_sa_w, rgb_sa_b, SaR);
  k_sa_conv7<<<gs, 256, 0, stream>>>(PoolC, chm_sa_w, chm_sa_b, SaC);
  k_channel_attn<<<BATCH, 256, 0, stream>>>(Orgb, rgb_fc1_w, rgb_fc1_b, rgb_fc2_w, rgb_fc2_b, CaR);
  k_channel_attn<<<BATCH, 256, 0, stream>>>(Ochm, chm_fc1_w, chm_fc1_b, chm_fc2_w, chm_fc2_b, CaC);

  // 7) final cross combine
  k_final<<<(int)((BCN + 255) / 256), 256, 0, stream>>>(Orgb, Ochm, SaR, SaC, CaR, CaC, out);
}
